// LlamaNaiveAdapter_4612794876119
// MI455X (gfx1250) — matrix operand order repaired and checked by reading, not compiled, not run on hardware
//
#include <hip/hip_runtime.h>
#include <math.h>

typedef __bf16 bf16;
typedef __attribute__((ext_vector_type(16))) __bf16 bf16x16;
typedef __attribute__((ext_vector_type(8)))  __bf16 bf16x8;
typedef __attribute__((ext_vector_type(8)))  float  f32x8;
typedef __attribute__((__vector_size__(4 * sizeof(int)))) int v4i;

#define SEQ   2048
#define HID   4096
#define NH    32
#define HD    128
#define PLEN  10
#define PPAD  32

// Async global->LDS copy (CDNA5), guarded so either toolchain compiles.
#if defined(__has_builtin)
#if __has_builtin(__builtin_amdgcn_global_load_async_to_lds_b128)
#define HAVE_ASYNC_LDS 1
#endif
#endif
#ifndef HAVE_ASYNC_LDS
#define HAVE_ASYNC_LDS 0
#endif

// ---------------- WMMA helpers (CDNA5 16x16x32 bf16, wave32) ----------------

__device__ __forceinline__ f32x8 wmma_bf16(bf16x16 a, bf16x16 b, f32x8 c) {
  // (neg_a, A, neg_b, B, c_mod, C, reuse_a, reuse_b)
  return __builtin_amdgcn_wmma_f32_16x16x32_bf16(false, a, false, b, (short)0, c,
                                                 false, false);
}

// A-matrix 16x32 bf16 fragment. Lane: m = lane%16, half = lane/16.
// element e (0..15): K = 16*half + e  -> 16 contiguous bf16.
__device__ __forceinline__ bf16x16 load_a_frag(const bf16* row_k0, int half) {
  bf16x8 lo = *reinterpret_cast<const bf16x8*>(row_k0 + 16 * half);
  bf16x8 hi = *reinterpret_cast<const bf16x8*>(row_k0 + 16 * half + 8);
  bf16x16 a;
#pragma unroll
  for (int e = 0; e < 8; ++e) { a[e] = lo[e]; a[e + 8] = hi[e]; }
  return a;
}

// B-matrix 32x16 bf16 fragment. Lane: n = lane%16, half = lane/16.
// element e (0..15): K = 16*half + e  -> 16 contiguous bf16.
__device__ __forceinline__ bf16x16 load_b_frag(const bf16* p) {
  bf16x8 lo = *reinterpret_cast<const bf16x8*>(p);
  bf16x8 hi = *reinterpret_cast<const bf16x8*>(p + 8);
  bf16x16 b;
#pragma unroll
  for (int e = 0; e < 8; ++e) { b[e] = lo[e]; b[e + 8] = hi[e]; }
  return b;
}

// B fragment sourced from f32 memory (weights), converted in-register.
__device__ __forceinline__ bf16x16 load_b_frag_f32(const float* p) {
  const float4* q = reinterpret_cast<const float4*>(p);
  float4 v0 = q[0], v1 = q[1], v2 = q[2], v3 = q[3];
  bf16x16 b;
  b[0]  = (bf16)v0.x; b[1]  = (bf16)v0.y; b[2]  = (bf16)v0.z; b[3]  = (bf16)v0.w;
  b[4]  = (bf16)v1.x; b[5]  = (bf16)v1.y; b[6]  = (bf16)v1.z; b[7]  = (bf16)v1.w;
  b[8]  = (bf16)v2.x; b[9]  = (bf16)v2.y; b[10] = (bf16)v2.z; b[11] = (bf16)v2.w;
  b[12] = (bf16)v3.x; b[13] = (bf16)v3.y; b[14] = (bf16)v3.z; b[15] = (bf16)v3.w;
  return b;
}

// ---------------- elementwise kernels ----------------

__global__ void k_cast_bf16(const float* __restrict__ in, bf16* __restrict__ out, int n) {
  int i = blockIdx.x * blockDim.x + threadIdx.x;
  if (i < n) out[i] = (bf16)in[i];
}

__global__ void k_pad_adapter(const float* __restrict__ in, bf16* __restrict__ out) {
  int i = blockIdx.x * blockDim.x + threadIdx.x;
  if (i >= PPAD * HID) return;
  int r = i / HID, c = i % HID;
  out[i] = (r < PLEN) ? (bf16)in[r * HID + c] : (bf16)0.0f;
}

// RoPE + reorder: [S][H] f32 -> bf16 [NH][S][HD]
__global__ void k_rope_reorder(const float* __restrict__ in, const int* __restrict__ pos,
                               bf16* __restrict__ out) {
  int i = blockIdx.x * blockDim.x + threadIdx.x;
  if (i >= SEQ * HID) return;
  int s = i / HID, o = i % HID;
  int h = o / HD, d = o % HD;
  int j = (d < 64) ? d : d - 64;
  float inv = __expf(-((float)(2 * j) / 128.0f) * 9.210340371976184f);  // 10000^(-2j/128)
  float ang = (float)pos[s] * inv;
  float c = __cosf(ang), sn = __sinf(ang);
  float x = in[(size_t)s * HID + o];
  float other = (d < 64) ? -in[(size_t)s * HID + h * HD + d + 64]
                         :  in[(size_t)s * HID + h * HD + d - 64];
  out[((size_t)h * SEQ + s) * HD + d] = (bf16)(x * c + other * sn);
}

// V [S][H] f32 -> Vt bf16 [NH][HD][SEQ]
__global__ void k_vt_reorder(const float* __restrict__ in, bf16* __restrict__ out) {
  int i = blockIdx.x * blockDim.x + threadIdx.x;
  if (i >= SEQ * HID) return;
  int s = i / HID, o = i % HID;
  int h = o / HD, d = o % HD;
  out[((size_t)h * HD + d) * SEQ + s] = (bf16)in[(size_t)s * HID + o];
}

// prefix K/V -> Kp bf16 [NH][PPAD][HD], Vpt bf16 [NH][HD][PPAD]
__global__ void k_prefix_reorder(const float* __restrict__ kp, const float* __restrict__ vp,
                                 bf16* __restrict__ Kp, bf16* __restrict__ Vpt) {
  int i = blockIdx.x * blockDim.x + threadIdx.x;
  if (i >= PPAD * HID) return;
  int r = i / HID, o = i % HID;
  int h = o / HD, d = o % HD;
  float kv = (r < PLEN) ? kp[(size_t)r * HID + o] : 0.0f;
  float vv = (r < PLEN) ? vp[(size_t)r * HID + o] : 0.0f;
  Kp[((size_t)h * PPAD + r) * HD + d] = (bf16)kv;
  Vpt[((size_t)h * HD + d) * PPAD + r] = (bf16)vv;
}

// ---- small GEMM (prefix, M=32): C = A[M,K]bf16 x W[N,K]f32^T, direct loads ----

__global__ void k_gemm_small(const bf16* __restrict__ A, const float* __restrict__ B,
                             float* __restrict__ C, int M, int N, int K,
                             int Mvalid, int nt64) {
  int wave = threadIdx.x >> 5;
  int lane = threadIdx.x & 31;
  int gw = blockIdx.x * 8 + wave;
  int mt = gw / nt64, nt = gw % nt64;
  int m0 = mt * 16, n0 = nt * 64;
  int m = lane & 15, half = lane >> 4;

  f32x8 acc[4];
#pragma unroll
  for (int t = 0; t < 4; ++t) { f32x8 z = {0,0,0,0,0,0,0,0}; acc[t] = z; }

  const bf16* arow = A + (size_t)(m0 + m) * K;
  for (int k0 = 0; k0 < K; k0 += 32) {
    bf16x16 a = load_a_frag(arow + k0, half);
#pragma unroll
    for (int t = 0; t < 4; ++t) {
      const float* pb = B + (size_t)(n0 + t * 16 + m) * K + k0 + 16 * half;
      acc[t] = wmma_bf16(a, load_b_frag_f32(pb), acc[t]);
    }
  }
#pragma unroll
  for (int t = 0; t < 4; ++t) {
    int col = n0 + t * 16 + m;
#pragma unroll
    for (int i = 0; i < 8; ++i) {
      int r = m0 + i + 8 * half;
      if (r < Mvalid) C[(size_t)r * N + col] = acc[t][i];
    }
  }
}

// ---- big GEMM: block computes 64x256, LDS-staged tiles, 8 waves ----
// A tile 64x32 bf16 (4KB) staged with async-to-LDS when available.
// B tile 256x32 staged f32->bf16 (16KB) once per block per k-step.

__global__ void k_gemm_big(const bf16* __restrict__ A, const float* __restrict__ B,
                           float* __restrict__ C, int M, int N, int K) {
  __shared__ bf16 sA[64 * 32];    // [row][k]
  __shared__ bf16 sB[256 * 32];   // [col][k]

  int tid = threadIdx.x;
  int wave = tid >> 5, lane = tid & 31;
  int m = lane & 15, half = lane >> 4;
  int mw = wave >> 2;          // 0..1  -> 32-row group
  int nw = wave & 3;           // 0..3  -> 64-col group

  int nblkN = N >> 8;          // N/256
  int bm = blockIdx.x / nblkN, bn = blockIdx.x % nblkN;
  int m0 = bm * 64, n0 = bn * 256;

  f32x8 acc0[4], acc1[4];
#pragma unroll
  for (int t = 0; t < 4; ++t) {
    f32x8 z = {0,0,0,0,0,0,0,0};
    acc0[t] = z; acc1[t] = z;
  }

  // per-thread staging coordinates
  int arow = tid >> 2, achk = tid & 3;            // A: 64 rows x 4 chunks of 16B
  const bf16* gA = A + (size_t)(m0 + arow) * K + achk * 8;
  bf16* lA = sA + arow * 32 + achk * 8;
  const float* gB = B + (size_t)(n0 + tid) * K;   // B: one 256-col row per thread
  bf16* lB = sB + tid * 32;

  for (int k0 = 0; k0 < K; k0 += 32) {
    __syncthreads();   // previous iteration's readers done before overwrite

    // ---- stage A (bf16 raw copy; async path if toolchain exposes it) ----
#if HAVE_ASYNC_LDS
    __builtin_amdgcn_global_load_async_to_lds_b128(
        (v4i*)(gA + k0), (v4i*)lA, 0, 0);
#else
    *reinterpret_cast<uint4*>(lA) = *reinterpret_cast<const uint4*>(gA + k0);
#endif

    // ---- stage B with f32 -> bf16 conversion ----
    const float4* gb4 = reinterpret_cast<const float4*>(gB + k0);
    __builtin_prefetch(gB + k0 + 64, 0, 0);
#pragma unroll
    for (int c = 0; c < 4; ++c) {
      float4 x = gb4[2 * c], y = gb4[2 * c + 1];
      bf16x8 v;
      v[0] = (bf16)x.x; v[1] = (bf16)x.y; v[2] = (bf16)x.z; v[3] = (bf16)x.w;
      v[4] = (bf16)y.x; v[5] = (bf16)y.y; v[6] = (bf16)y.z; v[7] = (bf16)y.w;
      *reinterpret_cast<bf16x8*>(lB + c * 8) = v;
    }

#if HAVE_ASYNC_LDS
    asm volatile("s_wait_asynccnt 0" ::: "memory");
#endif
    __syncthreads();

    // ---- compute: 2 A-frags x 4 B-frags = 8 WMMAs ----
    bf16x16 a0 = load_a_frag(sA + (mw * 32 + m) * 32, half);
    bf16x16 a1 = load_a_frag(sA + (mw * 32 + 16 + m) * 32, half);
#pragma unroll
    for (int t = 0; t < 4; ++t) {
      bf16x16 b = load_b_frag(sB + (nw * 64 + t * 16 + m) * 32 + 16 * half);
      acc0[t] = wmma_bf16(a0, b, acc0[t]);
      acc1[t] = wmma_bf16(a1, b, acc1[t]);
    }
  }

#pragma unroll
  for (int t = 0; t < 4; ++t) {
    int col = n0 + nw * 64 + t * 16 + m;
#pragma unroll
    for (int i = 0; i < 8; ++i) {
      int r = m0 + mw * 32 + i + 8 * half;
      C[(size_t)r * N + col] = acc0[t][i];
      C[(size_t)(r + 16) * N + col] = acc1[t][i];
    }
  }
}

// ---------------- flash attention w/ gated adapter prefix ----------------

__global__ void k_flash_attn(const bf16* __restrict__ Qh, const bf16* __restrict__ Kh,
                             const bf16* __restrict__ Vt, const bf16* __restrict__ Kp,
                             const bf16* __restrict__ Vpt, const float* __restrict__ gate,
                             bf16* __restrict__ Out) {
  __shared__ float lds[8][16 * 32];
  int wave = threadIdx.x >> 5;
  int lane = threadIdx.x & 31;
  int gw = blockIdx.x * 8 + wave;
  int head = gw >> 7;
  int qt = gw & 127;
  int q0 = qt * 16;
  int m = lane & 15, half = lane >> 4;

  const float scale = 0.08838834764831845f;  // 1/sqrt(128)
  const float NEG = -3.0e38f;

  const bf16* Qbase = Qh + ((size_t)head * SEQ + q0) * HD;
  bf16x16 aQ0 = load_a_frag(Qbase + (size_t)m * HD + 0,  half);
  bf16x16 aQ1 = load_a_frag(Qbase + (size_t)m * HD + 32, half);
  bf16x16 aQ2 = load_a_frag(Qbase + (size_t)m * HD + 64, half);
  bf16x16 aQ3 = load_a_frag(Qbase + (size_t)m * HD + 96, half);

  f32x8 O[8];
  float rowmax[8], rowsum[8];
#pragma unroll
  for (int t = 0; t < 8; ++t) { f32x8 z = {0,0,0,0,0,0,0,0}; O[t] = z; }
#pragma unroll
  for (int i = 0; i < 8; ++i) { rowmax[i] = NEG; rowsum[i] = 0.0f; }

  const bf16* Kbase = Kh + (size_t)head * SEQ * HD;
  const bf16* Vbase = Vt + (size_t)head * HD * SEQ;

  int nchunks = ((q0 + 15) >> 5) + 1;
  for (int c = 0; c < nchunks; ++c) {
    int j0 = c * 32;
    f32x8 s0 = {0,0,0,0,0,0,0,0};
    f32x8 s1 = {0,0,0,0,0,0,0,0};
#pragma unroll
    for (int kc = 0; kc < 4; ++kc) {
      bf16x16 a = (kc == 0) ? aQ0 : (kc == 1) ? aQ1 : (kc == 2) ? aQ2 : aQ3;
      bf16x16 b0 = load_b_frag(Kbase + (size_t)(j0 + m) * HD + kc * 32 + 16 * half);
      s0 = wmma_bf16(a, b0, s0);
      bf16x16 b1 = load_b_frag(Kbase + (size_t)(j0 + 16 + m) * HD + kc * 32 + 16 * half);
      s1 = wmma_bf16(a, b1, s1);
    }
#pragma unroll
    for (int i = 0; i < 8; ++i) {
      int r = q0 + i + 8 * half;
      float v0 = s0[i] * scale; if (j0 + m > r)      v0 = NEG;
      float v1 = s1[i] * scale; if (j0 + 16 + m > r) v1 = NEG;
      float mx = fmaxf(v0, v1);
      mx = fmaxf(mx, __shfl_xor(mx, 1, 32));
      mx = fmaxf(mx, __shfl_xor(mx, 2, 32));
      mx = fmaxf(mx, __shfl_xor(mx, 4, 32));
      mx = fmaxf(mx, __shfl_xor(mx, 8, 32));
      float nm = fmaxf(rowmax[i], mx);
      float corr = __expf(rowmax[i] - nm);
      rowmax[i] = nm;
      float p0 = __expf(v0 - nm);
      float p1 = __expf(v1 - nm);
      float rs = p0 + p1;
      rs += __shfl_xor(rs, 1, 32);
      rs += __shfl_xor(rs, 2, 32);
      rs += __shfl_xor(rs, 4, 32);
      rs += __shfl_xor(rs, 8, 32);
      rowsum[i] = rowsum[i] * corr + rs;
#pragma unroll
      for (int t = 0; t < 8; ++t) O[t][i] *= corr;
      lds[wave][(i + 8 * half) * 32 + m] = p0;
      lds[wave][(i + 8 * half) * 32 + 16 + m] = p1;
    }
    asm volatile("s_wait_dscnt 0" ::: "memory");
    bf16x16 aP;
#pragma unroll
    for (int e = 0; e < 8; ++e) {
      aP[e]     = (bf16)lds[wave][m * 32 + 16 * half + e];
      aP[e + 8] = (bf16)lds[wave][m * 32 + 16 * half + 8 + e];
    }
#pragma unroll
    for (int t = 0; t < 8; ++t) {
      bf16x16 bV = load_b_frag(Vbase + (size_t)(t * 16 + m) * SEQ + j0 + 16 * half);
      O[t] = wmma_bf16(aP, bV, O[t]);
    }
  }

  // ---- adapter prefix ----
  const bf16* Kpb = Kp + (size_t)head * PPAD * HD;
  const bf16* Vpb = Vpt + (size_t)head * HD * PPAD;
  f32x8 sp = {0,0,0,0,0,0,0,0};
#pragma unroll
  for (int kc = 0; kc < 4; ++kc) {
    bf16x16 a = (kc == 0) ? aQ0 : (kc == 1) ? aQ1 : (kc == 2) ? aQ2 : aQ3;
    bf16x16 b = load_b_frag(Kpb + (size_t)m * HD + kc * 32 + 16 * half);
    sp = wmma_bf16(a, b, sp);
  }
  float psum[8];
#pragma unroll
  for (int i = 0; i < 8; ++i) {
    float v = sp[i] * scale;
    if (m >= PLEN) v = NEG;
    float mx = v;
    mx = fmaxf(mx, __shfl_xor(mx, 1, 32));
    mx = fmaxf(mx, __shfl_xor(mx, 2, 32));
    mx = fmaxf(mx, __shfl_xor(mx, 4, 32));
    mx = fmaxf(mx, __shfl_xor(mx, 8, 32));
    float p = __expf(v - mx);
    float rs = p;
    rs += __shfl_xor(rs, 1, 32);
    rs += __shfl_xor(rs, 2, 32);
    rs += __shfl_xor(rs, 4, 32);
    rs += __shfl_xor(rs, 8, 32);
    psum[i] = rs;
    lds[wave][(i + 8 * half) * 32 + m] = p;
    lds[wave][(i + 8 * half) * 32 + 16 + m] = 0.0f;
  }
  asm volatile("s_wait_dscnt 0" ::: "memory");
  bf16x16 aP;
#pragma unroll
  for (int e = 0; e < 8; ++e) {
    aP[e]     = (bf16)lds[wave][m * 32 + 16 * half + e];
    aP[e + 8] = (bf16)lds[wave][m * 32 + 16 * half + 8 + e];
  }
  float g = tanhf(gate[head]);

#pragma unroll
  for (int t = 0; t < 8; ++t) {
    f32x8 z = {0,0,0,0,0,0,0,0};
    bf16x16 bV = load_b_frag(Vpb + (size_t)(t * 16 + m) * PPAD + 16 * half);
    f32x8 op = wmma_bf16(aP, bV, z);
#pragma unroll
    for (int i = 0; i < 8; ++i) {
      int r = q0 + i + 8 * half;
      float val = O[t][i] / rowsum[i] + g * op[i] / psum[i];
      Out[(size_t)r * HID + head * HD + t * 16 + m] = (bf16)val;
    }
  }
}

// ---------------- host launcher ----------------

extern "C" void kernel_launch(void* const* d_in, const int* in_sizes, int n_in,
                              void* d_out, int out_size, void* d_ws, size_t ws_size,
                              hipStream_t stream) {
  (void)in_sizes; (void)n_in; (void)out_size; (void)ws_size;
  const float* hs      = (const float*)d_in[0];
  const int*   pos     = (const int*)d_in[2];
  const float* Wq      = (const float*)d_in[3];
  const float* Wk      = (const float*)d_in[4];
  const float* Wv      = (const float*)d_in[5];
  const float* Wo      = (const float*)d_in[6];
  const float* adapter = (const float*)d_in[7];
  const float* gate    = (const float*)d_in[8];
  float* outp = (float*)d_out;

  char* w = (char*)d_ws;
  size_t off = 0;
  auto alloc = [&](size_t bytes) { char* p = w + off; off += (bytes + 255) & ~(size_t)255; return p; };

  bf16*  Xb   = (bf16*) alloc((size_t)SEQ * HID * 2);
  bf16*  Ab   = (bf16*) alloc((size_t)PPAD * HID * 2);
  float* Qf   = (float*)alloc((size_t)SEQ * HID * 4);
  float* Kf   = (float*)alloc((size_t)SEQ * HID * 4);
  float* Vf   = (float*)alloc((size_t)SEQ * HID * 4);
  float* Kpf  = (float*)alloc((size_t)PPAD * HID * 4);
  float* Vpf  = (float*)alloc((size_t)PPAD * HID * 4);
  bf16*  Qh   = (bf16*) alloc((size_t)SEQ * HID * 2);
  bf16*  Kh   = (bf16*) alloc((size_t)SEQ * HID * 2);
  bf16*  Vt   = (bf16*) alloc((size_t)SEQ * HID * 2);
  bf16*  Kp   = (bf16*) alloc((size_t)NH * PPAD * HD * 2);
  bf16*  Vpt  = (bf16*) alloc((size_t)NH * HD * PPAD * 2);
  bf16*  Attn = (bf16*) alloc((size_t)SEQ * HID * 2);

  const int TPB = 256;
  int nXH = SEQ * HID;
  int nAd = PPAD * HID;

  k_cast_bf16   <<<nXH / TPB, TPB, 0, stream>>>(hs, Xb, nXH);
  k_pad_adapter <<<nAd / TPB, TPB, 0, stream>>>(adapter, Ab);

  // big GEMMs: 64x256 per block
  int blocksBig = (SEQ / 64) * (HID / 256);   // 512
  k_gemm_big<<<blocksBig, TPB, 0, stream>>>(Xb, Wq, Qf, SEQ, HID, HID);
  k_gemm_big<<<blocksBig, TPB, 0, stream>>>(Xb, Wk, Kf, SEQ, HID, HID);
  k_gemm_big<<<blocksBig, TPB, 0, stream>>>(Xb, Wv, Vf, SEQ, HID, HID);

  // prefix GEMMs (M=32): direct kernel
  int nt64 = HID / 64;
  int blocksPre = ((PPAD / 16) * nt64) / 8;   // 16
  k_gemm_small<<<blocksPre, TPB, 0, stream>>>(Ab, Wk, Kpf, PPAD, HID, HID, PPAD, nt64);
  k_gemm_small<<<blocksPre, TPB, 0, stream>>>(Ab, Wv, Vpf, PPAD, HID, HID, PPAD, nt64);

  k_rope_reorder  <<<nXH / TPB, TPB, 0, stream>>>(Qf, pos, Qh);
  k_rope_reorder  <<<nXH / TPB, TPB, 0, stream>>>(Kf, pos, Kh);
  k_vt_reorder    <<<nXH / TPB, TPB, 0, stream>>>(Vf, Vt);
  k_prefix_reorder<<<nAd / TPB, TPB, 0, stream>>>(Kpf, Vpf, Kp, Vpt);

  int flashBlocks = (NH * (SEQ / 16)) / 8;    // 512
  k_flash_attn<<<flashBlocks, TPB, 0, stream>>>(Qh, Kh, Vt, Kp, Vpt, gate, Attn);

  k_gemm_big<<<blocksBig, TPB, 0, stream>>>(Attn, Wo, outp, SEQ, HID, HID);
}
